// Decoder_1382979469380
// MI455X (gfx1250) — matrix operand order repaired and checked by reading, not compiled, not run on hardware
//
#include <hip/hip_runtime.h>

#define DEV_INLINE __device__ __forceinline__

typedef __attribute__((ext_vector_type(16))) _Float16 v16h;
typedef __attribute__((ext_vector_type(8)))  _Float16 v8h;
typedef __attribute__((ext_vector_type(8)))  float    v8f;
typedef __attribute__((ext_vector_type(4)))  int      v4i;

static constexpr int B  = 32;
static constexpr int T  = 128;
static constexpr int H  = 1024;
static constexpr int L  = 2;
static constexpr int V  = 32000;
static constexpr int H3 = 3 * H;
static constexpr int BOS_IDX = 1;

// ---------------------------------------------------------------------------
// CDNA5 async global->LDS copy (ASYNCcnt-tracked), with sync fallback.
// Builtin signature (per clang diagnostic): (v4i AS1*, v4i AS3*, Imm, Imm).
// ---------------------------------------------------------------------------
#if defined(__gfx1250__) && __has_builtin(__builtin_amdgcn_global_load_async_to_lds_b128)
#define HAVE_ASYNC_LDS 1
#else
#define HAVE_ASYNC_LDS 0
#if defined(__gfx1250__)
#pragma message("gfx1250: async-to-LDS builtin NOT available; using sync LDS staging")
#endif
#endif

// Copy 16 bytes (8 halfs) from global to LDS, per lane.
DEV_INLINE void async_copy16(const _Float16* g, _Float16* l) {
#if HAVE_ASYNC_LDS
  typedef __attribute__((address_space(1))) v4i gv4i_t;
  typedef __attribute__((address_space(3))) v4i lv4i_t;
  __builtin_amdgcn_global_load_async_to_lds_b128(
      (gv4i_t*)g, (lv4i_t*)l, /*offset=*/0, /*cpol=*/0);
#else
  *(v8h*)l = *(const v8h*)g;
#endif
}

DEV_INLINE void tile_barrier() {
#if HAVE_ASYNC_LDS
  asm volatile("s_wait_asynccnt 0x0" ::: "memory");
#endif
  __syncthreads();
}

// ---------------------------------------------------------------------------
// WMMA helpers (wave32, 16x16x32 f16 -> f32). K order: contiguous, k = 16*(lane/16) + i, for A and B.
// ---------------------------------------------------------------------------
DEV_INLINE v8f wmma_f16(v16h a, v16h b, v8f c) {
  return __builtin_amdgcn_wmma_f32_16x16x32_f16(
      /*neg_a=*/false, a, /*neg_b=*/false, b,
      /*c_mod=*/(short)0, c, /*reuse_a=*/false, /*reuse_b=*/false);
}

// A fragment 16(M)x32(K) from row-major [rows x lda] f16 matrix.
// lanes 0-15: row = lane,    K = k0+[0..15]
// lanes 16-31: row = lane-16, K = k0+[16..31]
DEV_INLINE v16h load_a_frag(const _Float16* A, int lda, int row0, int k0) {
  int lane = threadIdx.x & 31;
  const _Float16* p = A + (size_t)(row0 + (lane & 15)) * lda + k0 + ((lane >> 4) << 4);
  v16h r;
#pragma unroll
  for (int i = 0; i < 16; ++i) r[i] = p[i];
  return r;
}

// B fragment 32(K)x16(N) where the weight matrix is stored [N x ldw] row-major
// (i.e. we compute X @ W^T).  lanes 0-15: N=lane, K=k0+[0..15];
// lanes 16-31: N=lane-16, K=k0+[16..31].
DEV_INLINE v16h load_b_frag(const _Float16* W, int ldw, int n0, int k0) {
  int lane = threadIdx.x & 31;
  const _Float16* p = W + (size_t)(n0 + (lane & 15)) * ldw + k0 + ((lane >> 4) << 4);
  v16h r;
#pragma unroll
  for (int i = 0; i < 16; ++i) r[i] = p[i];
  return r;
}

// ---------------------------------------------------------------------------
// Elementwise conversion / copy kernels
// ---------------------------------------------------------------------------
__global__ void k_f32_to_f16(const float* __restrict__ src,
                             _Float16* __restrict__ dst, long n) {
  long i = (long)blockIdx.x * blockDim.x + threadIdx.x;
  long stride = (long)gridDim.x * blockDim.x;
  for (; i < n; i += stride) dst[i] = (_Float16)src[i];
}

__global__ void k_copy_f32(const float* __restrict__ src,
                           float* __restrict__ dst, int n) {
  int i = blockIdx.x * blockDim.x + threadIdx.x;
  if (i < n) dst[i] = src[i];
}

// ---------------------------------------------------------------------------
// Teacher-forced embedding gather: ids = [BOS, target[:, :-1]]
// x16 layout: [T][B][H] (f16) -> rows of the gi0 GEMM
// ---------------------------------------------------------------------------
__global__ void k_embed(const int* __restrict__ tgt,
                        const float* __restrict__ emb,
                        _Float16* __restrict__ x16) {
  int idx = blockIdx.x;          // t*B + b
  int t = idx / B, b = idx % B;
  int id = (t == 0) ? BOS_IDX : tgt[b * T + (t - 1)];
  const float* src = emb + (size_t)id * H;
  _Float16* dst = x16 + (size_t)idx * H;
  for (int e = threadIdx.x; e < H; e += blockDim.x) dst[e] = (_Float16)src[e];
}

// ---------------------------------------------------------------------------
// Double-buffered LDS WMMA GEMM:
//   D[M,N] = A[M,K](f16) @ W[N,K]^T(f16) + bias  (f32 out)
// Block: 256 threads (8 waves), tile 128x128, BK=32.
// Tile k+1 streams into the other LDS buffer via GLOBAL_LOAD_ASYNC_TO_LDS
// while WMMAs consume tile k; one s_wait_asynccnt + barrier per K-step.
// wave grid 4(M) x 2(N): each wave computes 2x4 = 8 16x16 tiles.
// ---------------------------------------------------------------------------
static constexpr int GEMM_LD = 40;  // LDS row stride in halfs (pad vs 64 banks)

__global__ void __launch_bounds__(256)
k_gemm_f16(const _Float16* __restrict__ A, const _Float16* __restrict__ W,
           const float* __restrict__ bias, float* __restrict__ D,
           int M, int N, int K, int ldd) {
  __shared__ _Float16 sA[2][128 * GEMM_LD];
  __shared__ _Float16 sB[2][128 * GEMM_LD];
  (void)M; (void)N;

  const int m0 = blockIdx.y * 128;
  const int n0 = blockIdx.x * 128;
  const int tid = threadIdx.x;
  const int wid = tid >> 5;
  const int waveM = wid & 3;       // 0..3
  const int waveN = wid >> 2;      // 0..1

  const int lr = tid >> 1;         // 0..127 : LDS row this thread fills
  const int lc = (tid & 1) * 16;   // 0 or 16

  const _Float16* gA = A + (size_t)(m0 + lr) * K + lc;
  const _Float16* gB = W + (size_t)(n0 + lr) * K + lc;
  const int lofs = lr * GEMM_LD + lc;

  auto issue_tile = [&](int buf, int k0) {
    async_copy16(gA + k0,     &sA[buf][lofs]);
    async_copy16(gA + k0 + 8, &sA[buf][lofs + 8]);
    async_copy16(gB + k0,     &sB[buf][lofs]);
    async_copy16(gB + k0 + 8, &sB[buf][lofs + 8]);
  };

  v8f acc[2][4];
#pragma unroll
  for (int mi = 0; mi < 2; ++mi)
#pragma unroll
    for (int ni = 0; ni < 4; ++ni) acc[mi][ni] = {};

  issue_tile(0, 0);
  tile_barrier();

  int buf = 0;
  for (int k0 = 0; k0 < K; k0 += 32) {
    if (k0 + 32 < K) issue_tile(buf ^ 1, k0 + 32);

    v16h af[2];
#pragma unroll
    for (int mi = 0; mi < 2; ++mi)
      af[mi] = load_a_frag(sA[buf], GEMM_LD, waveM * 32 + mi * 16, 0);
    v16h bf[4];
#pragma unroll
    for (int ni = 0; ni < 4; ++ni)
      bf[ni] = load_b_frag(sB[buf], GEMM_LD, waveN * 64 + ni * 16, 0);
#pragma unroll
    for (int mi = 0; mi < 2; ++mi)
#pragma unroll
      for (int ni = 0; ni < 4; ++ni)
        acc[mi][ni] = wmma_f16(af[mi], bf[ni], acc[mi][ni]);

    tile_barrier();
    buf ^= 1;
  }

  const int lane = tid & 31;
#pragma unroll
  for (int mi = 0; mi < 2; ++mi) {
#pragma unroll
    for (int ni = 0; ni < 4; ++ni) {
      int m = m0 + waveM * 32 + mi * 16 + ((lane >> 4) << 3);
      int n = n0 + waveN * 64 + ni * 16 + (lane & 15);
      float bv = bias[n];
      float* p = D + (size_t)m * ldd + n;
#pragma unroll
      for (int i = 0; i < 8; ++i) p[(size_t)i * ldd] = acc[mi][ni][i] + bv;
    }
  }
}

// ---------------------------------------------------------------------------
// One GRU step for one layer. grid = (B/16, H/16), 32 threads (1 wave).
// Wave computes the r/z/n gate columns [j, j+H, j+2H) for 16 batch rows.
// LAYER==0: gi is precomputed (gi0 GEMM, bias included); compute gh via WMMA.
// LAYER==1: x = layer-0 output of this step; compute gi and gh via WMMA.
// h state is double-buffered (prev -> next) to avoid intra-launch races.
// ---------------------------------------------------------------------------
template <int LAYER>
__global__ void __launch_bounds__(32)
k_gru_step(int t,
           const _Float16* __restrict__ h16_prev,  // [L][B][H]
           _Float16* __restrict__ h16_next,        // [L][B][H]
           const float* __restrict__ h32_prev,     // [L][B][H]
           float* __restrict__ h32_next,           // [L][B][H]
           const _Float16* __restrict__ whh16,     // [L][3H][H]
           const _Float16* __restrict__ wih16,     // [L][3H][H]
           const float* __restrict__ bhh,          // [L][3H]
           const float* __restrict__ bih,          // [L][3H]
           const float* __restrict__ gi0,          // [T*B][3H]
           _Float16* __restrict__ hs16) {          // [B*T][H]
  const int bm = blockIdx.x;   // batch tile
  const int jn = blockIdx.y;   // hidden tile
  const int lane = threadIdx.x;

  const _Float16* hA  = h16_prev + (size_t)LAYER * B * H;
  const _Float16* xA  = h16_next;  // layer-0 slice of the *current* step
  const _Float16* Whh = whh16 + (size_t)LAYER * H3 * H;
  const _Float16* Wih = wih16 + (size_t)LAYER * H3 * H;

  if (LAYER == 0)  // warm the gi0 rows consumed in the epilogue
    __builtin_prefetch(gi0 + ((size_t)t * B + bm * 16) * H3 + jn * 16, 0, 1);

  v8f ghr = {}, ghz = {}, ghn = {};
  v8f gir = {}, giz = {}, gin = {};

  for (int k0 = 0; k0 < H; k0 += 32) {
    // hint the next K-chunk of the weight streams into cache
    if (k0 + 32 < H) {
      __builtin_prefetch(Whh + (size_t)(jn * 16 + (lane & 15)) * H + k0 + 32, 0, 3);
      if (LAYER == 1)
        __builtin_prefetch(Wih + (size_t)(jn * 16 + (lane & 15)) * H + k0 + 32, 0, 3);
    }
    v16h a = load_a_frag(hA, H, bm * 16, k0);
    ghr = wmma_f16(a, load_b_frag(Whh, H, jn * 16,          k0), ghr);
    ghz = wmma_f16(a, load_b_frag(Whh, H, H + jn * 16,      k0), ghz);
    ghn = wmma_f16(a, load_b_frag(Whh, H, 2 * H + jn * 16,  k0), ghn);
    if (LAYER == 1) {
      v16h ax = load_a_frag(xA, H, bm * 16, k0);
      gir = wmma_f16(ax, load_b_frag(Wih, H, jn * 16,         k0), gir);
      giz = wmma_f16(ax, load_b_frag(Wih, H, H + jn * 16,     k0), giz);
      gin = wmma_f16(ax, load_b_frag(Wih, H, 2 * H + jn * 16, k0), gin);
    }
  }

  const int n  = jn * 16 + (lane & 15);
  const int mb = bm * 16 + ((lane >> 4) << 3);
  const float* bh = bhh + LAYER * H3;
  const float* bi = bih + LAYER * H3;
  const float bhr = bh[n], bhz = bh[H + n], bhn = bh[2 * H + n];
  const float bir = (LAYER == 1) ? bi[n]         : 0.0f;
  const float biz = (LAYER == 1) ? bi[H + n]     : 0.0f;
  const float bin = (LAYER == 1) ? bi[2 * H + n] : 0.0f;

#pragma unroll
  for (int i = 0; i < 8; ++i) {
    const int m = mb + i;  // batch index
    float gr, gz, gn_;
    if (LAYER == 0) {
      const float* g = gi0 + ((size_t)t * B + m) * H3;
      gr = g[n]; gz = g[H + n]; gn_ = g[2 * H + n];
    } else {
      gr = gir[i] + bir; gz = giz[i] + biz; gn_ = gin[i] + bin;
    }
    const float hr = ghr[i] + bhr, hz = ghz[i] + bhz, hn = ghn[i] + bhn;
    const float r  = 1.0f / (1.0f + __expf(-(gr + hr)));
    const float z  = 1.0f / (1.0f + __expf(-(gz + hz)));
    const float nn = tanhf(gn_ + r * hn);
    const size_t hidx = (size_t)LAYER * B * H + (size_t)m * H + n;
    const float hv = (1.0f - z) * nn + z * h32_prev[hidx];
    h32_next[hidx] = hv;
    h16_next[hidx] = (_Float16)hv;
    if (LAYER == 1) hs16[((size_t)m * T + t) * H + n] = (_Float16)hv;
  }
}

// ---------------------------------------------------------------------------
// In-place log-softmax over V for each of B*T rows.
// ---------------------------------------------------------------------------
__global__ void __launch_bounds__(256)
k_log_softmax(float* __restrict__ logits) {
  __shared__ float red[256];
  float* p = logits + (size_t)blockIdx.x * V;
  const int tid = threadIdx.x;

  float mx = -3.0e38f;
  for (int i = tid; i < V; i += 256) mx = fmaxf(mx, p[i]);
  red[tid] = mx; __syncthreads();
  for (int s = 128; s > 0; s >>= 1) {
    if (tid < s) red[tid] = fmaxf(red[tid], red[tid + s]);
    __syncthreads();
  }
  mx = red[0]; __syncthreads();

  float sum = 0.0f;
  for (int i = tid; i < V; i += 256) sum += __expf(p[i] - mx);
  red[tid] = sum; __syncthreads();
  for (int s = 128; s > 0; s >>= 1) {
    if (tid < s) red[tid] += red[tid + s];
    __syncthreads();
  }
  const float lse = mx + __logf(red[0]);

  for (int i = tid; i < V; i += 256) p[i] = p[i] - lse;
}

// ---------------------------------------------------------------------------
extern "C" void kernel_launch(void* const* d_in, const int* in_sizes, int n_in,
                              void* d_out, int out_size, void* d_ws, size_t ws_size,
                              hipStream_t stream) {
  (void)in_sizes; (void)n_in; (void)out_size; (void)ws_size;
  const float* enc_hid = (const float*)d_in[1];
  const int*   target  = (const int*)d_in[2];
  const float* emb     = (const float*)d_in[3];
  const float* w_ih    = (const float*)d_in[4];
  const float* w_hh    = (const float*)d_in[5];
  const float* b_ih    = (const float*)d_in[6];
  const float* b_hh    = (const float*)d_in[7];
  const float* w_out   = (const float*)d_in[8];
  const float* b_out   = (const float*)d_in[9];
  float* out = (float*)d_out;

  // Workspace carve-out
  char* ws = (char*)d_ws;
  size_t off = 0;
  auto carve = [&](size_t bytes) {
    void* p = ws + off;
    off = (off + bytes + 255) & ~(size_t)255;
    return p;
  };
  _Float16* wih16 = (_Float16*)carve((size_t)L * H3 * H * sizeof(_Float16));
  _Float16* whh16 = (_Float16*)carve((size_t)L * H3 * H * sizeof(_Float16));
  _Float16* wout16 = (_Float16*)carve((size_t)V * H * sizeof(_Float16));
  _Float16* x16    = (_Float16*)carve((size_t)T * B * H * sizeof(_Float16));
  _Float16* hs16   = (_Float16*)carve((size_t)B * T * H * sizeof(_Float16));
  float*    gi0    = (float*)carve((size_t)T * B * H3 * sizeof(float));
  float*    h32a   = (float*)carve((size_t)L * B * H * sizeof(float));
  float*    h32b   = (float*)carve((size_t)L * B * H * sizeof(float));
  _Float16* h16a   = (_Float16*)carve((size_t)L * B * H * sizeof(_Float16));
  _Float16* h16b   = (_Float16*)carve((size_t)L * B * H * sizeof(_Float16));

  // 1) f32 -> f16 weight conversions (L2-resident afterwards)
  k_f32_to_f16<<<2048, 256, 0, stream>>>(w_ih, wih16, (long)L * H3 * H);
  k_f32_to_f16<<<2048, 256, 0, stream>>>(w_hh, whh16, (long)L * H3 * H);
  k_f32_to_f16<<<8192, 256, 0, stream>>>(w_out, wout16, (long)V * H);
  k_f32_to_f16<<<256, 256, 0, stream>>>(enc_hid, h16a, (long)L * B * H);
  (void)hipMemcpyAsync(h32a, enc_hid, (size_t)L * B * H * sizeof(float),
                       hipMemcpyDeviceToDevice, stream);

  // 2) teacher-forced embedding gather -> x16 [T][B][H]
  k_embed<<<T * B, 256, 0, stream>>>(target, emb, x16);

  // 3) hoisted layer-0 input transform: gi0 = x @ w_ih0^T + b_ih0  [T*B, 3H]
  k_gemm_f16<<<dim3(H3 / 128, (T * B) / 128), 256, 0, stream>>>(
      x16, wih16, b_ih, gi0, T * B, H3, H, H3);

  // 4) sequential GRU scan with ping-pong h state
  float*    h32c = h32a; float*    h32n = h32b;
  _Float16* h16c = h16a; _Float16* h16n = h16b;
  const dim3 gruGrid(B / 16, H / 16);
  for (int t = 0; t < T; ++t) {
    k_gru_step<0><<<gruGrid, 32, 0, stream>>>(
        t, h16c, h16n, h32c, h32n, whh16, wih16, b_hh, b_ih, gi0, hs16);
    k_gru_step<1><<<gruGrid, 32, 0, stream>>>(
        t, h16c, h16n, h32c, h32n, whh16, wih16, b_hh, b_ih, gi0, hs16);
    float* tf = h32c; h32c = h32n; h32n = tf;
    _Float16* th = h16c; h16c = h16n; h16n = th;
  }

  // 5) output projection: logits[B*T, V] = hs @ w_out^T + b_out (into d_out)
  k_gemm_f16<<<dim3(V / 128, (B * T) / 128), 256, 0, stream>>>(
      hs16, wout16, b_out, out, B * T, V, H, V);

  // 6) in-place log-softmax per row
  k_log_softmax<<<B * T, 256, 0, stream>>>(out);

  // 7) h_final -> tail of d_out
  k_copy_f32<<<(L * B * H + 255) / 256, 256, 0, stream>>>(
      h32c, out + (size_t)B * T * V, L * B * H);
}
